// SelfAttention_70239895158862
// MI455X (gfx1250) — matrix operand order repaired and checked by reading, not compiled, not run on hardware
//
#include <hip/hip_runtime.h>

// ---------------------------------------------------------------------------
// Problem constants (match reference)
// ---------------------------------------------------------------------------
constexpr int Bc   = 8;
constexpr int Nc   = 4096;
constexpr int Cc   = 1280;
constexpr int Dqk  = 160;
constexpr int OUTc = 1024;
constexpr float LARGEc = 1000000.0f;

// ---------------------------------------------------------------------------
// WMMA types / helpers (CDNA5 gfx1250, wave32)
// ---------------------------------------------------------------------------
typedef __attribute__((ext_vector_type(16))) __bf16        v16bf;
typedef __attribute__((ext_vector_type(8)))  float         v8f;
typedef __attribute__((ext_vector_type(4)))  unsigned int  v4u;
typedef __attribute__((ext_vector_type(4)))  int           v4i;
typedef __attribute__((ext_vector_type(8)))  int           v8i;

union Frag {
    v16bf        bf;
    unsigned int u32[8];
};

__device__ inline unsigned short f32_to_bf16(float f) {
    unsigned int u = __builtin_bit_cast(unsigned int, f);
    u += 0x7FFFu + ((u >> 16) & 1u);          // round-to-nearest-even
    return (unsigned short)(u >> 16);
}

__device__ inline v8f wmma_bf16(const Frag& a, const Frag& b, v8f c) {
    // D(16x16 f32) = A(16x32 bf16) * B(32x16 bf16) + C
    return __builtin_amdgcn_wmma_f32_16x16x32_bf16(
        false, a.bf, false, b.bf, (short)0, c, false, false);
}

// A-matrix 16x32 bf16 lane layout: lane m = lane&15, hi = lane>>4
//   VGPR v holds K pair starting at: 16*hi + 2v (contiguous, matches B)
__device__ inline int a_frag_kbase(int v, int hi) {
    return 16 * hi + 2 * v;
}
// B 32x16 bf16, this code's k order: lane n = lane&15; lanes 0-15 hold K 0..15, lanes 16-31
// hold K 16..31; VGPR v holds pair (koff + 2v, koff + 2v + 1), koff = 16*hi.

// ---------------------------------------------------------------------------
// Tensor Data Mover support (gfx1250): DMA a 2-D bf16 tile Global -> LDS
// ---------------------------------------------------------------------------
#if defined(__has_builtin)
#  if __has_builtin(__builtin_amdgcn_tensor_load_to_lds) && \
      __has_builtin(__builtin_amdgcn_s_wait_tensorcnt)
#    define HAVE_TDM 1
#  endif
#endif
#ifndef HAVE_TDM
#  define HAVE_TDM 0
#endif

#if HAVE_TDM
// D# per CDNA5 ISA: tile_rows x tile_cols (bf16) from a
// row-major tensor (row stride = tensor_d0_stride elements) into LDS at
// lds_byte_off.
__device__ inline void tdm_load_tile_bf16(
    const void* gbase, unsigned int lds_byte_off,
    unsigned int tensor_d0, unsigned int tensor_d1, unsigned int d0_stride,
    unsigned int tile_d0, unsigned int tile_d1)
{
    const unsigned long long ga = (unsigned long long)gbase;
    v4u g0 = { 1u,                                   // count=1, load, user mode
               lds_byte_off,                         // lds_addr
               (unsigned int)(ga & 0xFFFFFFFFull),   // global_addr[31:0]
               (unsigned int)((ga >> 32) & 0x01FFFFFFull) | (2u << 30) }; // type=2
    v8i g1 = { (int)(1u << 16),                      // data_size=1 (2 bytes)
               (int)(tensor_d0 << 16),               // tensor_dim0[15:0]
               (int)(tensor_d1 << 16),               // dim0[31:16]=0 | tensor_dim1[15:0]
               (int)(tile_d0 << 16),                 // dim1[31:16]=0 | tile_dim0
               (int)tile_d1,                         // tile_dim1 | tile_dim2=0
               (int)d0_stride,                       // tensor_dim0_stride[31:0]
               0, 0 };
    v4i g2 = { 0, 0, 0, 0 };
    v4i g3 = { 0, 0, 0, 0 };
#if __clang_major__ >= 23
    v8i gz = { 0, 0, 0, 0, 0, 0, 0, 0 };
    __builtin_amdgcn_tensor_load_to_lds(g0, g1, g2, g3, gz, 0);
#else
    __builtin_amdgcn_tensor_load_to_lds(g0, g1, g2, g3, 0);
#endif
}
#endif

// ===========================================================================
// Kernel 1/4: tiled GEMM  out(M x Nout) = A(M x K) * W(Nout x K)^T + bias
//   A: fp32 or bf16(raw u16);  out: bf16(raw u16) or fp32
//   Block 256 thr (8 waves), tile 64(M) x 32(N), K-step 64 (2 WMMAs/step).
// ===========================================================================
template <bool A_BF16, bool OUT_BF16>
__global__ __launch_bounds__(256) void gemm_wmma(
    const void* __restrict__ Aptr, const float* __restrict__ W,
    const float* __restrict__ bias, void* __restrict__ outp,
    int M, int K, int Nout)
{
    __shared__ unsigned short As[64 * 64];   // [m][k]   8 KB
    __shared__ unsigned short Bs[32 * 64];   // [n][k]   4 KB

    const int tid  = threadIdx.x;
    const int lane = tid & 31;
    const int wave = tid >> 5;
    const int wm   = wave & 3;      // 4 M-subtiles
    const int wn   = wave >> 2;     // 2 N-subtiles
    const int m0   = blockIdx.x * 64;
    const int n0   = blockIdx.y * 32;
    const int ml   = lane & 15;
    const int hi   = lane >> 4;

    v8f acc = {};

    for (int kk = 0; kk < K; kk += 64) {
        // ---- stage A tile (64x64) as bf16 ----
        if (A_BF16) {
            const unsigned short* A = (const unsigned short*)Aptr;
            for (int i = tid; i < 64 * 32; i += 256) {         // dwords
                int row = i >> 5, c2 = (i & 31) << 1;
                unsigned int d = *(const unsigned int*)&A[(size_t)(m0 + row) * K + kk + c2];
                *(unsigned int*)&As[row * 64 + c2] = d;
            }
        } else {
            const float* A = (const float*)Aptr;
            for (int i = tid; i < 64 * 64; i += 256) {
                int row = i >> 6, c = i & 63;
                As[row * 64 + c] = f32_to_bf16(A[(size_t)(m0 + row) * K + kk + c]);
            }
        }
        // ---- stage W tile (32 n x 64 k) as bf16 ----
        for (int i = tid; i < 32 * 64; i += 256) {
            int n = i >> 6, c = i & 63;
            Bs[n * 64 + c] = f32_to_bf16(W[(size_t)(n0 + n) * K + kk + c]);
        }
        __syncthreads();

        Frag a0, a1, b0, b1;
#pragma unroll
        for (int v = 0; v < 8; ++v) {
            const int ak = a_frag_kbase(v, hi);
            const int bk = hi * 16 + 2 * v;
            a0.u32[v] = *(const unsigned int*)&As[(wm * 16 + ml) * 64 + ak];
            a1.u32[v] = *(const unsigned int*)&As[(wm * 16 + ml) * 64 + 32 + ak];
            b0.u32[v] = *(const unsigned int*)&Bs[(wn * 16 + ml) * 64 + bk];
            b1.u32[v] = *(const unsigned int*)&Bs[(wn * 16 + ml) * 64 + 32 + bk];
        }
        acc = wmma_bf16(a0, b0, acc);
        acc = wmma_bf16(a1, b1, acc);
        __syncthreads();
    }

    // epilogue: D layout row = r + 8*hi, col = lane&15
    const int col = n0 + wn * 16 + ml;
    const float bval = bias[col];
#pragma unroll
    for (int r = 0; r < 8; ++r) {
        const int row = m0 + wm * 16 + r + 8 * hi;
        const float val = acc[r] + bval;
        if (OUT_BF16)
            ((unsigned short*)outp)[(size_t)row * Nout + col] = f32_to_bf16(val);
        else
            ((float*)outp)[(size_t)row * Nout + col] = val;
    }
}

// ===========================================================================
// Kernel 2: attention pass 1 — per-row softmax stats (max, sumexp)
//   8 independent waves per block; each wave owns one 16-query tile.
//   Bulk-load all 10 K-fragments per 32-key tile, then 10 WMMAs back-to-back.
// ===========================================================================
__global__ __launch_bounds__(256) void attn_stats(
    const unsigned short* __restrict__ Qb, const unsigned short* __restrict__ Kb,
    const float* __restrict__ mask, float* __restrict__ rowmax,
    float* __restrict__ rowsum)
{
    const int tid  = threadIdx.x;
    const int lane = tid & 31;
    const int wave = tid >> 5;
    const int qt   = blockIdx.x * 8 + wave;   // global 16-query tile id
    const int b    = qt >> 8;                 // N/16 = 256 tiles per batch
    const int q0   = (qt & 255) << 4;
    const int ml   = lane & 15;
    const int hi   = lane >> 4;

    // Q fragments: 5 chunks of 16x32 (Dqk = 160)
    Frag qf[5];
#pragma unroll
    for (int c = 0; c < 5; ++c)
#pragma unroll
        for (int v = 0; v < 8; ++v)
            qf[c].u32[v] = *(const unsigned int*)
                &Qb[(size_t)(b * Nc + q0 + ml) * Dqk + c * 32 + a_frag_kbase(v, hi)];

    float mrun[8], lrun[8];
#pragma unroll
    for (int r = 0; r < 8; ++r) { mrun[r] = -3.0e38f; lrun[r] = 0.0f; }

    for (int kb0 = 0; kb0 < Nc; kb0 += 32) {
        // bulk-load 10 B-fragments (keys kb0..kb0+31, all 5 d-chunks)
        const unsigned short* k0 = &Kb[(size_t)(b * Nc + kb0 + ml) * Dqk];
        const unsigned short* k1 = &Kb[(size_t)(b * Nc + kb0 + 16 + ml) * Dqk];
        Frag kf[10];
#pragma unroll
        for (int c = 0; c < 5; ++c)
#pragma unroll
            for (int v = 0; v < 8; ++v) {
                const int koff = c * 32 + hi * 16 + 2 * v;
                kf[2 * c].u32[v]     = *(const unsigned int*)&k0[koff];
                kf[2 * c + 1].u32[v] = *(const unsigned int*)&k1[koff];
            }
        const float pen0 = (1.0f - mask[b * Nc + kb0 + ml]) * LARGEc;
        const float pen1 = (1.0f - mask[b * Nc + kb0 + 16 + ml]) * LARGEc;

        v8f s0 = {}, s1 = {};
#pragma unroll
        for (int c = 0; c < 5; ++c) {
            s0 = wmma_bf16(qf[c], kf[2 * c], s0);
            s1 = wmma_bf16(qf[c], kf[2 * c + 1], s1);
        }
#pragma unroll
        for (int r = 0; r < 8; ++r) {
            const float a0 = s0[r] - pen0;
            const float a1 = s1[r] - pen1;
            float tm = fmaxf(a0, a1);
            for (int off = 1; off < 16; off <<= 1)
                tm = fmaxf(tm, __shfl_xor(tm, off, 32));
            const float mnew = fmaxf(mrun[r], tm);
            float p = __expf(a0 - mnew) + __expf(a1 - mnew);
            for (int off = 1; off < 16; off <<= 1)
                p += __shfl_xor(p, off, 32);
            lrun[r] = lrun[r] * __expf(mrun[r] - mnew) + p;
            mrun[r] = mnew;
        }
    }
    if (ml == 0) {
#pragma unroll
        for (int r = 0; r < 8; ++r) {
            const int row = b * Nc + q0 + r + 8 * hi;
            rowmax[row] = mrun[r];
            rowsum[row] = lrun[r];
        }
    }
}

// ===========================================================================
// Kernel 3: attention pass 2 — out = softmax(S)·V, fused y = gamma*out + x
//   One block (8 waves) per 16-query tile. K tile DMA'd by the Tensor Data
//   Mover (TDM) while all threads transpose-stage the V tile; each wave owns
//   160 of the 1280 V columns (10 x 16x16 f32 accumulators).
// ===========================================================================
// LDS layout (bytes): sV [Cc][32] @0 (81920) | sK [32][Dqk] @81920 (10240)
//                     | sP 8*[16][32] @92160 (8192)  -> total 100352
constexpr unsigned int SK_LDS_OFF = (unsigned int)Cc * 32u * 2u;

__global__ __launch_bounds__(256) void attn_pv(
    const unsigned short* __restrict__ Qb, const unsigned short* __restrict__ Kb,
    const unsigned short* __restrict__ Vb, const float* __restrict__ mask,
    const float* __restrict__ rowmax, const float* __restrict__ rowsum,
    const float* __restrict__ x, const float* __restrict__ gamma,
    unsigned short* __restrict__ Yb)
{
    extern __shared__ unsigned short smem[];
    unsigned short* sV = smem;                       // [Cc][32] transposed V tile
    unsigned short* sK = smem + (size_t)Cc * 32;     // [32][Dqk]
    unsigned short* sP = sK + 32 * Dqk;              // 8 waves * [16][32]

    const int tid  = threadIdx.x;
    const int lane = tid & 31;
    const int wave = tid >> 5;
    const int b    = blockIdx.y;
    const int q0   = blockIdx.x << 4;
    const int ml   = lane & 15;
    const int hi   = lane >> 4;
    unsigned short* myP = sP + wave * 16 * 32;

    // Q fragments (resident)
    Frag qf[5];
#pragma unroll
    for (int c = 0; c < 5; ++c)
#pragma unroll
        for (int v = 0; v < 8; ++v)
            qf[c].u32[v] = *(const unsigned int*)
                &Qb[(size_t)(b * Nc + q0 + ml) * Dqk + c * 32 + a_frag_kbase(v, hi)];

    // softmax stats for my 8 rows
    float rm[8], rsi[8];
#pragma unroll
    for (int r = 0; r < 8; ++r) {
        const int row = b * Nc + q0 + r + 8 * hi;
        rm[r]  = rowmax[row];
        rsi[r] = 1.0f / rowsum[row];
    }

    const v8f vzero = {};
    v8f acc[10];
#pragma unroll
    for (int t = 0; t < 10; ++t) acc[t] = vzero;

    const float g = gamma[0];

    for (int kb0 = 0; kb0 < Nc; kb0 += 32) {
        // ---- stage K tile: TDM DMA (wave 0) or manual fallback ----
#if HAVE_TDM
        if (wave == 0) {
            tdm_load_tile_bf16(&Kb[(size_t)(b * Nc + kb0) * Dqk], SK_LDS_OFF,
                               /*tensor_d0=*/Dqk, /*tensor_d1=*/Nc,
                               /*d0_stride=*/Dqk, /*tile_d0=*/Dqk, /*tile_d1=*/32);
        }
#else
        {
            const unsigned int* src = (const unsigned int*)&Kb[(size_t)(b * Nc + kb0) * Dqk];
            unsigned int* dst = (unsigned int*)sK;
            for (int i = tid; i < (32 * Dqk) / 2; i += 256) dst[i] = src[i];
        }
#endif
        // ---- stage V tile transposed: sV[e*32 + k] (overlaps with TDM) ----
        {
            const unsigned int* src = (const unsigned int*)&Vb[(size_t)(b * Nc + kb0) * Cc];
            for (int i = tid; i < 32 * (Cc / 2); i += 256) {
                const int k  = i / (Cc / 2);
                const int e2 = i % (Cc / 2);
                const unsigned int d = src[(size_t)k * (Cc / 2) + e2];
                sV[(2 * e2) * 32 + k]     = (unsigned short)(d & 0xFFFFu);
                sV[(2 * e2 + 1) * 32 + k] = (unsigned short)(d >> 16);
            }
        }
#if HAVE_TDM
        if (wave == 0) __builtin_amdgcn_s_wait_tensorcnt((short)0);
#endif
        __syncthreads();

        // ---- S = Q * Ktile^T (16 x 32): bulk LDS fragment loads, then WMMAs
        Frag kf[10];
#pragma unroll
        for (int c = 0; c < 5; ++c)
#pragma unroll
            for (int v = 0; v < 8; ++v) {
                const int koff = c * 32 + hi * 16 + 2 * v;
                kf[2 * c].u32[v]     = *(const unsigned int*)&sK[ml * Dqk + koff];
                kf[2 * c + 1].u32[v] = *(const unsigned int*)&sK[(16 + ml) * Dqk + koff];
            }
        v8f s0 = vzero, s1 = vzero;
#pragma unroll
        for (int c = 0; c < 5; ++c) {
            s0 = wmma_bf16(qf[c], kf[2 * c], s0);
            s1 = wmma_bf16(qf[c], kf[2 * c + 1], s1);
        }
        const float pen0 = (1.0f - mask[b * Nc + kb0 + ml]) * LARGEc;
        const float pen1 = (1.0f - mask[b * Nc + kb0 + 16 + ml]) * LARGEc;

        // ---- P = exp(S - rowmax) / rowsum, D-layout -> LDS (bf16) ----
#pragma unroll
        for (int r = 0; r < 8; ++r) {
            const int row = r + 8 * hi;
            myP[row * 32 + ml]      = f32_to_bf16(__expf(s0[r] - pen0 - rm[r]) * rsi[r]);
            myP[row * 32 + 16 + ml] = f32_to_bf16(__expf(s1[r] - pen1 - rm[r]) * rsi[r]);
        }
        // ---- reload P in A-matrix layout ----
        Frag pf;
#pragma unroll
        for (int v = 0; v < 8; ++v)
            pf.u32[v] = *(const unsigned int*)&myP[ml * 32 + a_frag_kbase(v, hi)];

        // ---- acc += P(16x32) * Vtile(32 x my 160 cols), prefetch depth 2 ----
        Frag vcur, vnxt;
#pragma unroll
        for (int v = 0; v < 8; ++v)
            vcur.u32[v] = *(const unsigned int*)&sV[(wave * 160 + ml) * 32 + hi * 16 + 2 * v];
#pragma unroll
        for (int t = 0; t < 10; ++t) {
            if (t < 9) {
                const int e0 = wave * 160 + (t + 1) * 16;
#pragma unroll
                for (int v = 0; v < 8; ++v)
                    vnxt.u32[v] = *(const unsigned int*)&sV[(e0 + ml) * 32 + hi * 16 + 2 * v];
            }
            acc[t] = wmma_bf16(pf, vcur, acc[t]);
            vcur = vnxt;
        }
        __syncthreads();   // before restaging K/V
    }

    // ---- epilogue: y = gamma*out + x  -> bf16 ----
#pragma unroll
    for (int t = 0; t < 10; ++t) {
        const int e = wave * 160 + t * 16 + ml;
#pragma unroll
        for (int r = 0; r < 8; ++r) {
            const size_t idx = (size_t)(b * Nc + q0 + r + 8 * hi) * Cc + e;
            Yb[idx] = f32_to_bf16(g * acc[t][r] + x[idx]);
        }
    }
}

// ===========================================================================
// Host-side launch
// ===========================================================================
extern "C" void kernel_launch(void* const* d_in, const int* in_sizes, int n_in,
                              void* d_out, int out_size, void* d_ws, size_t ws_size,
                              hipStream_t stream)
{
    (void)in_sizes; (void)n_in; (void)out_size; (void)ws_size;

    const float* x     = (const float*)d_in[0];
    const float* mask  = (const float*)d_in[1];
    const float* Wq    = (const float*)d_in[2];
    const float* bq    = (const float*)d_in[3];
    const float* Wk    = (const float*)d_in[4];
    const float* bk    = (const float*)d_in[5];
    const float* Wv    = (const float*)d_in[6];
    const float* bv    = (const float*)d_in[7];
    const float* gamma = (const float*)d_in[8];
    const float* Wp    = (const float*)d_in[9];
    const float* bp    = (const float*)d_in[10];
    float*       out   = (float*)d_out;

    const size_t M = (size_t)Bc * Nc;   // 32768 token rows

    // workspace partition (bf16 stored as raw u16)
    char* ws = (char*)d_ws;
    size_t off = 0;
    unsigned short* Qbf = (unsigned short*)(ws + off); off += M * Dqk * 2;
    unsigned short* Kbf = (unsigned short*)(ws + off); off += M * Dqk * 2;
    unsigned short* Vbf = (unsigned short*)(ws + off); off += M * Cc * 2;
    unsigned short* Ybf = (unsigned short*)(ws + off); off += M * Cc * 2;
    float* rowmax = (float*)(ws + off); off += M * 4;
    float* rowsum = (float*)(ws + off); off += M * 4;

    dim3 blk(256);

    // Q = x*Wq^T + bq ; K = x*Wk^T + bk   (fp32 in, bf16 out)
    gemm_wmma<false, true><<<dim3((int)(M / 64), Dqk / 32), blk, 0, stream>>>(
        (const void*)x, Wq, bq, (void*)Qbf, (int)M, Cc, Dqk);
    gemm_wmma<false, true><<<dim3((int)(M / 64), Dqk / 32), blk, 0, stream>>>(
        (const void*)x, Wk, bk, (void*)Kbf, (int)M, Cc, Dqk);
    // V = x*Wv^T + bv
    gemm_wmma<false, true><<<dim3((int)(M / 64), Cc / 32), blk, 0, stream>>>(
        (const void*)x, Wv, bv, (void*)Vbf, (int)M, Cc, Cc);

    // pass 1: softmax row stats (8 q-tiles per block)
    attn_stats<<<dim3((Bc * (Nc / 16)) / 8), blk, 0, stream>>>(
        Qbf, Kbf, mask, rowmax, rowsum);

    // pass 2: P·V with fused gamma-residual -> Ybf (bf16)
    const size_t smem_bytes = ((size_t)Cc * 32 + 32 * Dqk + 8 * 16 * 32) * 2;  // ~98 KB
    attn_pv<<<dim3(Nc / 16, Bc), blk, smem_bytes, stream>>>(
        Qbf, Kbf, Vbf, mask, rowmax, rowsum, x, gamma, Ybf);

    // final projection: out = y*Wp^T + bp  (bf16 in, fp32 out)
    gemm_wmma<true, false><<<dim3((int)(M / 64), OUTc / 32), blk, 0, stream>>>(
        (const void*)Ybf, Wp, bp, (void*)out, (int)M, Cc, OUTc);
}
